// LSTMModel_75866302316565
// MI455X (gfx1250) — compile-verified, matrix operand order checked by reading, not run on hardware
//
#include <hip/hip_runtime.h>
#include <hip/hip_bf16.h>
#include <math.h>

// Problem constants (match reference).
#define B_  32
#define T_  2048
#define D_  1024
#define H_  256     // 4H = 1024, 2H = 512

typedef __bf16 bf16;
typedef __bf16 v16bf __attribute__((ext_vector_type(16)));
typedef __bf16 bfx8  __attribute__((ext_vector_type(8)));
typedef float  v8f   __attribute__((ext_vector_type(8)));

// ---- Tensor Data Mover availability / arity detection ----------------------
#if defined(__has_builtin)
#  if __has_builtin(__builtin_amdgcn_tensor_load_to_lds)
#    define HAVE_TDM 1
#  else
#    define HAVE_TDM 0
#  endif
#else
#  define HAVE_TDM 0
#endif

#if __has_include(<hip/amd_detail/amd_gfx1250_TDM.h>)
#  define TDM_6ARG 1   // therock-10.0 headers => clang-23 6-arg builtin
#else
#  define TDM_6ARG 0   // ROCm 7.2 clang-22 5-arg builtin
#endif

// Build a 16-element bf16 WMMA fragment from two contiguous 16B chunks.
// Per CDNA5 ISA (16-bit A/B fragment): VGPRs 0-3 hold K run [k0, k0+8),
// VGPRs 4-7 hold [k0+16, k0+24), with k0 = ktile*32 + (lane/16)*8.
static __device__ __forceinline__ v16bf mkfrag(bfx8 lo, bfx8 hi) {
  v16bf r;
#pragma unroll
  for (int i = 0; i < 8; ++i) { r[i] = lo[i]; r[i + 8] = hi[i]; }
  return r;
}

static __device__ __forceinline__ v8f wmma_bf16(v16bf a, v16bf b, v8f c) {
  // (neg_a, A, neg_b, B, c_mod, C, reuse_a, reuse_b)
  return __builtin_amdgcn_wmma_f32_16x16x32_bf16(false, a, false, b,
                                                 (short)0, c, false, false);
}

static __device__ __forceinline__ float sigm(float x) {
  return 1.0f / (1.0f + __expf(-x));
}

#if HAVE_TDM
typedef unsigned int u32x4 __attribute__((ext_vector_type(4)));
typedef int          i32x4 __attribute__((ext_vector_type(4)));
typedef int          i32x8 __attribute__((ext_vector_type(8)));

// Issue one TDM tile load: 128 rows x 32 bf16 cols from a row-major matrix
// with row stride `strideElems` (bf16 elements), into LDS laid out as
// [128][40] bf16 (64B data + 16B pad per row, produced by the TDM pad unit).
// D# field packing per CDNA5 ISA 8.3-8.5. Groups 2/3 unused (2D tile).
static __device__ __forceinline__ void tdm_load_tile(
    unsigned lds_off, const bf16* gsrc, unsigned strideElems)
{
  const unsigned long long ga = (unsigned long long)(uintptr_t)gsrc;
  u32x4 g0;
  g0[0] = 1u;                                   // count=1: one valid user D#
  g0[1] = lds_off;                              // lds_addr (bytes)
  g0[2] = (unsigned)ga;                         // global_addr[31:0]
  g0[3] = (unsigned)((ga >> 32) & 0x01ffffffu)  // global_addr[56:32]
          | (2u << 30);                         // type = 2 ("image")
  i32x8 g1;
  g1[0] = (int)((1u << 16)     // data_size = 1 -> 2-byte elements
              | (1u << 20)     // pad_enable (LDS destination padding)
              | (3u << 22)     // pad_interval: 16 DWORDs (one 64B row)
              | (3u << 25));   // pad_amount:   4 DWORDs (16B) per interval
  g1[1] = (int)(32u  << 16);   // tensor_dim0[15:0]=32 at bits[79:64-16]...
  g1[2] = (int)(128u << 16);   // tensor_dim0[31:16]=0 | tensor_dim1[15:0]=128
  g1[3] = (int)(32u  << 16);   // tensor_dim1[31:16]=0 | tile_dim0=32
  g1[4] = (int)128u;           // tile_dim1=128 | tile_dim2=0
  g1[5] = (int)strideElems;    // tensor_dim0_stride[31:0]
  g1[6] = 0;                   // stride0[47:32] | stride1[15:0] (unused, 2D)
  g1[7] = 0;                   // stride1[47:16]
  const i32x4 gz = {0, 0, 0, 0};
#if TDM_6ARG
  const i32x8 gz8 = {0, 0, 0, 0, 0, 0, 0, 0};
  __builtin_amdgcn_tensor_load_to_lds(g0, g1, gz, gz, gz8, 0);
#else
  __builtin_amdgcn_tensor_load_to_lds(g0, g1, gz, gz, 0);
#endif
}
#endif  // HAVE_TDM

// ---------------------------------------------------------------------------
// K0: f32 -> bf16 conversion (with NaN->0 for x, matching jnp.nan_to_num).
// Pure streaming: non-temporal loads, plain stores.
// ---------------------------------------------------------------------------
__global__ void cvt_f32_bf16(const float* __restrict__ src, bf16* __restrict__ dst,
                             size_t n, int zero_nan) {
  size_t i = (size_t)blockIdx.x * blockDim.x + threadIdx.x;
  size_t stride = (size_t)gridDim.x * blockDim.x;
  for (; i < n; i += stride) {
    float v = __builtin_nontemporal_load(src + i);
    if (zero_nan && v != v) v = 0.0f;
    dst[i] = (bf16)v;
  }
}

// ---------------------------------------------------------------------------
// K1/K3: C[m, coff+n] = sum_k A[m,k] * W[n,k] + bias[n]
// A: [M,K] bf16 row-major, W: [N,K] bf16 row-major (i.e. B = W^T), C f32.
// Block tile 128x128, K-step 32, 256 threads = 8 waves, each wave 4x2 tiles.
// Staging: TDM (tensor_load_to_lds) double-buffered when available;
// otherwise manual global->LDS copies.
// ---------------------------------------------------------------------------
__global__ __launch_bounds__(256) void wmma_gemm_bias(
    const bf16* __restrict__ A, const bf16* __restrict__ W,
    const float* __restrict__ bias, float* __restrict__ C,
    int M, int N, int K, int ldc, int coff)
{
  (void)M;
#if HAVE_TDM
  __shared__ __align__(16) bf16 As[2][128][40];   // double-buffered, TDM-padded
  __shared__ __align__(16) bf16 Ws[2][128][40];
#else
  __shared__ __align__(16) bf16 As1[128][40];
  __shared__ __align__(16) bf16 Ws1[128][40];
#endif

  const int tid  = threadIdx.x;
  const int lane = tid & 31;
  const int wave = tid >> 5;
  const int half = lane >> 4;
  const int l16  = lane & 15;
  const int wm = (wave & 1) * 64;   // wave m-offset in block tile
  const int wn = (wave >> 1) * 32;  // wave n-offset in block tile
  const int bm = blockIdx.y * 128;
  const int bn = blockIdx.x * 128;

  const v8f zero8 = {0.f, 0.f, 0.f, 0.f, 0.f, 0.f, 0.f, 0.f};
  v8f acc[4][2];
#pragma unroll
  for (int mt = 0; mt < 4; ++mt)
#pragma unroll
    for (int nt = 0; nt < 2; ++nt) acc[mt][nt] = zero8;

  const int nk = K / 32;

#if HAVE_TDM
  // Prime the pipeline: one wave drives the Tensor Data Mover for the
  // whole workgroup (TDM ignores EXEC; one instruction per wave).
  if (wave == 0) {
    tdm_load_tile((unsigned)(uintptr_t)&As[0][0][0], A + (size_t)bm * K, (unsigned)K);
    tdm_load_tile((unsigned)(uintptr_t)&Ws[0][0][0], W + (size_t)bn * K, (unsigned)K);
  }
#else
  const int lrow = tid >> 2;          // 0..63, two passes cover 128 rows
  const int lchk = (tid & 3) * 8;     // bf16 offset of 16B chunk: 0,8,16,24
#endif

  for (int kk = 0; kk < nk; ++kk) {
#if HAVE_TDM
    const int cur = kk & 1;
    if (wave == 0) {
      if (kk + 1 < nk) {
        const size_t knext = (size_t)(kk + 1) * 32;
        tdm_load_tile((unsigned)(uintptr_t)&As[cur ^ 1][0][0],
                      A + (size_t)bm * K + knext, (unsigned)K);
        tdm_load_tile((unsigned)(uintptr_t)&Ws[cur ^ 1][0][0],
                      W + (size_t)bn * K + knext, (unsigned)K);
        __builtin_amdgcn_s_wait_tensorcnt(2);   // current pair complete
      } else {
        __builtin_amdgcn_s_wait_tensorcnt(0);   // drain on last step
      }
    }
    __syncthreads();                            // LDS tiles visible to all
    const bf16 (*Asb)[40] = As[cur];
    const bf16 (*Wsb)[40] = Ws[cur];
#else
    const int k0 = kk * 32;
    __syncthreads();
    *(bfx8*)&As1[lrow][lchk]      = *(const bfx8*)&A[(size_t)(bm + lrow)      * K + k0 + lchk];
    *(bfx8*)&As1[lrow + 64][lchk] = *(const bfx8*)&A[(size_t)(bm + lrow + 64) * K + k0 + lchk];
    *(bfx8*)&Ws1[lrow][lchk]      = *(const bfx8*)&W[(size_t)(bn + lrow)      * K + k0 + lchk];
    *(bfx8*)&Ws1[lrow + 64][lchk] = *(const bfx8*)&W[(size_t)(bn + lrow + 64) * K + k0 + lchk];
    __syncthreads();
    const bf16 (*Asb)[40] = As1;
    const bf16 (*Wsb)[40] = Ws1;
#endif

    v16bf bfr[2];
#pragma unroll
    for (int nt = 0; nt < 2; ++nt) {
      const int n = wn + nt * 16 + l16;
      bfr[nt] = mkfrag(*(const bfx8*)&Wsb[n][half * 8],
                       *(const bfx8*)&Wsb[n][16 + half * 8]);
    }
#pragma unroll
    for (int mt = 0; mt < 4; ++mt) {
      const int m = wm + mt * 16 + l16;
      v16bf afr = mkfrag(*(const bfx8*)&Asb[m][half * 8],
                         *(const bfx8*)&Asb[m][16 + half * 8]);
#pragma unroll
      for (int nt = 0; nt < 2; ++nt)
        acc[mt][nt] = wmma_bf16(afr, bfr[nt], acc[mt][nt]);
    }
    __syncthreads();                            // tiles consumed; safe to reuse
  }

  // Epilogue: D element (r, lane) sits at m = r + 8*half, n = lane&15.
  // C is stream-out (read once later): non-temporal stores keep L2 for Whh.
#pragma unroll
  for (int nt = 0; nt < 2; ++nt) {
    const int n = bn + wn + nt * 16 + l16;
    const float bv = bias ? bias[n] : 0.0f;
#pragma unroll
    for (int mt = 0; mt < 4; ++mt) {
      const int mbase = bm + wm + mt * 16 + half * 8;
#pragma unroll
      for (int r = 0; r < 8; ++r)
        __builtin_nontemporal_store(acc[mt][nt][r] + bv,
                                    &C[(size_t)(mbase + r) * ldc + coff + n]);
    }
  }
}

// ---------------------------------------------------------------------------
// K2/K4: bidirectional LSTM scan for one layer.
// grid.x = 2 (dir 0 = forward, dir 1 = backward), 512 threads = 16 waves.
// Precomputed G[b*T+t, 4H] = x@Wih^T + b (f32, HBM-streamed, NT + prefetch).
// Whh is [4H,256] bf16 and stays L2-resident (512KB << 192MB L2).
// Wave w owns hidden columns [w*16, w*16+16) across ALL FOUR gates
// (n-tiles w, w+16, w+32, w+48), so i/f/g/o for a given (batch, col) land in
// the same lane/accumulator slot -> gate math stays in registers.
// Cell state c lives in registers; h broadcast via a 17KB LDS buffer.
// ---------------------------------------------------------------------------
__global__ __launch_bounds__(512) void lstm_bidir(
    const float* __restrict__ Gf, const float* __restrict__ Gb,
    const bf16* __restrict__ Whhf, const bf16* __restrict__ Whhb,
    bf16* __restrict__ hout_bf, float* __restrict__ hout_f32,
    int write_bf16)
{
  const int dir = blockIdx.x;
  const float* __restrict__ G   = dir ? Gb   : Gf;
  const bf16*  __restrict__ Whh = dir ? Whhb : Whhf;
  const int hoff = dir ? H_ : 0;

  __shared__ __align__(16) bf16 hbuf[32][272];   // [batch][H + 16 pad]

  const int tid  = threadIdx.x;
  const int lane = tid & 31;
  const int wave = tid >> 5;
  const int half = lane >> 4;
  const int l16  = lane & 15;
  const int hc   = wave * 16 + l16;    // hidden column owned by this lane

  for (int i = tid; i < 32 * 272; i += 512) ((bf16*)hbuf)[i] = (bf16)0.0f;

  float c[2][8];
#pragma unroll
  for (int mt = 0; mt < 2; ++mt)
#pragma unroll
    for (int r = 0; r < 8; ++r) c[mt][r] = 0.0f;

  __syncthreads();

  const v8f zero8 = {0.f, 0.f, 0.f, 0.f, 0.f, 0.f, 0.f, 0.f};

  for (int t = 0; t < T_; ++t) {
    const int tp = dir ? (T_ - 1 - t) : t;

    // Prefetch next step's gate pre-activations into cache while the WMMA
    // phase runs (global_prefetch_b8 warms a full 64B line per 16 lanes).
    if (t + 1 < T_) {
      const int tq = dir ? (T_ - 2 - t) : (t + 1);
#pragma unroll
      for (int mt = 0; mt < 2; ++mt)
#pragma unroll
        for (int r = 0; r < 8; ++r) {
          const int b = mt * 16 + half * 8 + r;
          const float* gp = G + ((size_t)b * T_ + tq) * (4 * H_) + hc;
#pragma unroll
          for (int g = 0; g < 4; ++g) __builtin_prefetch(gp + g * H_, 0, 0);
        }
    }

    v8f acc[2][4];                       // [m-tile][gate]
#pragma unroll
    for (int mt = 0; mt < 2; ++mt)
#pragma unroll
      for (int g = 0; g < 4; ++g) acc[mt][g] = zero8;

#pragma unroll
    for (int kf = 0; kf < 8; ++kf) {     // K = 256 = 8 x 32
      const int k0 = kf * 32 + half * 8;
      v16bf a0 = mkfrag(*(const bfx8*)&hbuf[l16][k0],
                        *(const bfx8*)&hbuf[l16][k0 + 16]);
      v16bf a1 = mkfrag(*(const bfx8*)&hbuf[16 + l16][k0],
                        *(const bfx8*)&hbuf[16 + l16][k0 + 16]);
#pragma unroll
      for (int g = 0; g < 4; ++g) {
        const bf16* wrow = Whh + (size_t)(g * H_ + wave * 16 + l16) * H_;
        v16bf bw = mkfrag(*(const bfx8*)&wrow[k0], *(const bfx8*)&wrow[k0 + 16]);
        acc[0][g] = wmma_bf16(a0, bw, acc[0][g]);
        acc[1][g] = wmma_bf16(a1, bw, acc[1][g]);
      }
    }
    __syncthreads();   // everyone finished reading old h from hbuf

    // Gate nonlinearities fully in registers; write new h.
#pragma unroll
    for (int mt = 0; mt < 2; ++mt) {
#pragma unroll
      for (int r = 0; r < 8; ++r) {
        const int b = mt * 16 + half * 8 + r;                 // batch index
        const size_t grow = ((size_t)b * T_ + tp) * (4 * H_);
        const float zi = acc[mt][0][r] + __builtin_nontemporal_load(&G[grow + hc]);
        const float zf = acc[mt][1][r] + __builtin_nontemporal_load(&G[grow + H_ + hc]);
        const float zg = acc[mt][2][r] + __builtin_nontemporal_load(&G[grow + 2 * H_ + hc]);
        const float zo = acc[mt][3][r] + __builtin_nontemporal_load(&G[grow + 3 * H_ + hc]);
        const float cn = sigm(zf) * c[mt][r] + sigm(zi) * tanhf(zg);
        c[mt][r] = cn;
        const float h = sigm(zo) * tanhf(cn);
        hbuf[b][hc] = (bf16)h;
        const size_t orow = ((size_t)b * T_ + tp) * (2 * H_) + hoff + hc;
        if (write_bf16) {
          __builtin_nontemporal_store(
              __builtin_bit_cast(unsigned short, (bf16)h),
              (unsigned short*)hout_bf + orow);
        } else {
          __builtin_nontemporal_store(h, hout_f32 + orow);
        }
      }
    }
    __syncthreads();   // new h visible before next step's fragment loads
  }
}

// ---------------------------------------------------------------------------
// K5: z = h1 + res; LayerNorm(512); hid = relu(z@Wc1^T + bc1); logit.
// One 128-thread block per (b,t) row. h1/res streamed (NT); Wc1 L2-hot.
// ---------------------------------------------------------------------------
__global__ __launch_bounds__(128) void head_kernel(
    const float* __restrict__ h1, const float* __restrict__ res,
    const float* __restrict__ gamma, const float* __restrict__ beta,
    const float* __restrict__ Wc1, const float* __restrict__ bc1,
    const float* __restrict__ Wc2, const float* __restrict__ bc2,
    float* __restrict__ out)
{
  __shared__ __align__(16) float zrow[512];
  __shared__ float red[128];
  const int tid = threadIdx.x;
  const size_t row = blockIdx.x;
  const float* hr = h1 + row * 512;
  const float* rr = res + row * 512;

  float s = 0.0f;
#pragma unroll
  for (int j = 0; j < 4; ++j) {
    const int k = tid + 128 * j;
    const float v = __builtin_nontemporal_load(hr + k) +
                    __builtin_nontemporal_load(rr + k);
    zrow[k] = v;
    s += v;
  }
  red[tid] = s;
  __syncthreads();
#pragma unroll
  for (int off = 64; off > 0; off >>= 1) {
    if (tid < off) red[tid] += red[tid + off];
    __syncthreads();
  }
  const float mu = red[0] * (1.0f / 512.0f);
  __syncthreads();

  float v2 = 0.0f;
#pragma unroll
  for (int j = 0; j < 4; ++j) {
    const float d = zrow[tid + 128 * j] - mu;
    v2 += d * d;
  }
  red[tid] = v2;
  __syncthreads();
#pragma unroll
  for (int off = 64; off > 0; off >>= 1) {
    if (tid < off) red[tid] += red[tid + off];
    __syncthreads();
  }
  const float rstd = rsqrtf(red[0] * (1.0f / 512.0f) + 1e-5f);
  __syncthreads();

#pragma unroll
  for (int j = 0; j < 4; ++j) {
    const int k = tid + 128 * j;
    zrow[k] = (zrow[k] - mu) * rstd * gamma[k] + beta[k];
  }
  __syncthreads();

  const float4* wr  = (const float4*)(Wc1 + (size_t)tid * 512);
  const float4* zr4 = (const float4*)zrow;
  float acc = bc1[tid];
#pragma unroll 4
  for (int k4 = 0; k4 < 128; ++k4) {
    const float4 w = wr[k4];
    const float4 z = zr4[k4];
    acc += w.x * z.x + w.y * z.y + w.z * z.z + w.w * z.w;
  }
  const float hid = fmaxf(acc, 0.0f);
  red[tid] = hid * Wc2[tid];
  __syncthreads();
#pragma unroll
  for (int off = 64; off > 0; off >>= 1) {
    if (tid < off) red[tid] += red[tid + off];
    __syncthreads();
  }
  if (tid == 0) __builtin_nontemporal_store(red[0] + bc2[0], out + row);
}

// ---------------------------------------------------------------------------
// Host orchestration.
// ---------------------------------------------------------------------------
extern "C" void kernel_launch(void* const* d_in, const int* in_sizes, int n_in,
                              void* d_out, int out_size, void* d_ws, size_t ws_size,
                              hipStream_t stream) {
  (void)in_sizes; (void)n_in; (void)out_size; (void)ws_size;

  const float* x     = (const float*)d_in[0];
  const float* Wih0f = (const float*)d_in[1];
  const float* Whh0f = (const float*)d_in[2];
  const float* b0f   = (const float*)d_in[3];
  const float* Wih0b = (const float*)d_in[4];
  const float* Whh0b = (const float*)d_in[5];
  const float* b0b   = (const float*)d_in[6];
  const float* Wih1f = (const float*)d_in[7];
  const float* Whh1f = (const float*)d_in[8];
  const float* b1f   = (const float*)d_in[9];
  const float* Wih1b = (const float*)d_in[10];
  const float* Whh1b = (const float*)d_in[11];
  const float* b1b   = (const float*)d_in[12];
  const float* Wres  = (const float*)d_in[13];
  const float* bres  = (const float*)d_in[14];
  const float* gamma = (const float*)d_in[15];
  const float* beta  = (const float*)d_in[16];
  const float* Wc1   = (const float*)d_in[17];
  const float* bc1   = (const float*)d_in[18];
  const float* Wc2   = (const float*)d_in[19];
  const float* bc2   = (const float*)d_in[20];

  const size_t MT = (size_t)B_ * T_;   // 65536 rows

  // Workspace carve-up (~882 MB total).
  char* ws = (char*)d_ws;
  size_t off = 0;
  auto take = [&](size_t bytes) -> char* {
    char* p = ws + off;
    off += (bytes + 255) & ~(size_t)255;
    return p;
  };
  bf16* xbf      = (bf16*)take(MT * D_ * 2);          // reused as h1 (f32) later
  bf16* wih0f_bf = (bf16*)take((size_t)4 * H_ * D_ * 2);
  bf16* wih0b_bf = (bf16*)take((size_t)4 * H_ * D_ * 2);
  bf16* wih1f_bf = (bf16*)take((size_t)4 * H_ * 2 * H_ * 2);
  bf16* wih1b_bf = (bf16*)take((size_t)4 * H_ * 2 * H_ * 2);
  bf16* whh0f_bf = (bf16*)take((size_t)4 * H_ * H_ * 2);
  bf16* whh0b_bf = (bf16*)take((size_t)4 * H_ * H_ * 2);
  bf16* whh1f_bf = (bf16*)take((size_t)4 * H_ * H_ * 2);
  bf16* whh1b_bf = (bf16*)take((size_t)4 * H_ * H_ * 2);
  bf16* wres_bf  = (bf16*)take((size_t)2 * H_ * D_ * 2);
  float* Gf      = (float*)take(MT * 4 * H_ * 4);     // layer0 fwd, reused layer1 fwd
  float* Gb      = (float*)take(MT * 4 * H_ * 4);     // layer0 bwd, reused layer1 bwd
  float* resbuf  = (float*)take(MT * 2 * H_ * 4);
  bf16* h0bf     = (bf16*)take(MT * 2 * H_ * 2);
  float* h1      = (float*)xbf;                       // alias: x_bf16 dead after K1

  auto cvt = [&](const float* s, bf16* d, size_t n, int zn) {
    size_t want = (n + 255) / 256;
    int blocks = (int)(want < 8192 ? want : 8192);
    cvt_f32_bf16<<<dim3(blocks), dim3(256), 0, stream>>>(s, d, n, zn);
  };

  // --- K0: precision conversion (x gets nan_to_num) -------------------------
  cvt(x,     xbf,      MT * D_,                1);
  cvt(Wih0f, wih0f_bf, (size_t)4 * H_ * D_,     0);
  cvt(Wih0b, wih0b_bf, (size_t)4 * H_ * D_,     0);
  cvt(Wih1f, wih1f_bf, (size_t)4 * H_ * 2 * H_, 0);
  cvt(Wih1b, wih1b_bf, (size_t)4 * H_ * 2 * H_, 0);
  cvt(Whh0f, whh0f_bf, (size_t)4 * H_ * H_,     0);
  cvt(Whh0b, whh0b_bf, (size_t)4 * H_ * H_,     0);
  cvt(Whh1f, whh1f_bf, (size_t)4 * H_ * H_,     0);
  cvt(Whh1b, whh1b_bf, (size_t)4 * H_ * H_,     0);
  cvt(Wres,  wres_bf,  (size_t)2 * H_ * D_,     0);

  const dim3 blk256(256);
  const int MB = (int)(MT / 128);     // 512 block-rows

  // --- K1: input projections + residual (WMMA GEMMs, TDM-staged) ------------
  wmma_gemm_bias<<<dim3(1024 / 128, MB), blk256, 0, stream>>>(
      xbf, wih0f_bf, b0f, Gf, (int)MT, 1024, D_, 1024, 0);
  wmma_gemm_bias<<<dim3(1024 / 128, MB), blk256, 0, stream>>>(
      xbf, wih0b_bf, b0b, Gb, (int)MT, 1024, D_, 1024, 0);
  wmma_gemm_bias<<<dim3(512 / 128, MB), blk256, 0, stream>>>(
      xbf, wres_bf, bres, resbuf, (int)MT, 512, D_, 512, 0);

  // --- K2: layer-0 bidirectional scan (fwd+bwd concurrently, grid=2) --------
  lstm_bidir<<<dim3(2), dim3(512), 0, stream>>>(
      Gf, Gb, whh0f_bf, whh0b_bf, h0bf, (float*)nullptr, 1);

  // --- K3: layer-1 input projections (K = 2H = 512) -------------------------
  wmma_gemm_bias<<<dim3(1024 / 128, MB), blk256, 0, stream>>>(
      h0bf, wih1f_bf, b1f, Gf, (int)MT, 1024, 2 * H_, 1024, 0);
  wmma_gemm_bias<<<dim3(1024 / 128, MB), blk256, 0, stream>>>(
      h0bf, wih1b_bf, b1b, Gb, (int)MT, 1024, 2 * H_, 1024, 0);

  // --- K4: layer-1 bidirectional scan (writes f32 h1) -----------------------
  lstm_bidir<<<dim3(2), dim3(512), 0, stream>>>(
      Gf, Gb, whh1f_bf, whh1b_bf, (bf16*)nullptr, h1, 0);

  // --- K5: residual + LayerNorm + classifier head ---------------------------
  head_kernel<<<dim3((unsigned)MT), dim3(128), 0, stream>>>(
      h1, resbuf, gamma, beta, Wc1, bc1, Wc2, bc2, (float*)d_out);
}
